// MultiHeadAttention_30906584662328
// MI455X (gfx1250) — matrix operand order repaired and checked by reading, not compiled, not run on hardware
//
#include <hip/hip_runtime.h>

typedef __attribute__((ext_vector_type(16))) _Float16 v16h;
typedef __attribute__((ext_vector_type(8)))  _Float16 v8h;
typedef __attribute__((ext_vector_type(4)))  _Float16 v4h;
typedef __attribute__((ext_vector_type(8)))  float    v8f;

#define N_NODES  100000
#define IN_DIM   128
#define HID      128            // NUM_HEADS(8) * OUT_DIM(16)
#define N_EDGES  1600000

union AV  { _Float16 e[16]; v16h v; };
union BV  { v8h h8[2];      v16h v; };
union V4U { v4h v; _Float16 e[4]; };

// ---------------------------------------------------------------- zero output
__global__ void zero_out_kernel(float4* __restrict__ p, int n4) {
    int i = blockIdx.x * blockDim.x + threadIdx.x;
    if (i < n4) p[i] = make_float4(0.f, 0.f, 0.f, 0.f);
}

// ------------------------------------------- transpose+convert W -> f16 [n][k]
__global__ void prep_w_kernel(const float* __restrict__ Wq,
                              const float* __restrict__ Wk,
                              const float* __restrict__ Wv,
                              _Float16* __restrict__ Wt) {
    int i = blockIdx.x * blockDim.x + threadIdx.x;   // over 3*128*128
    if (i >= 3 * HID * IN_DIM) return;
    int m3 = i / (HID * IN_DIM);
    int r  = i % (HID * IN_DIM);
    int n  = r / IN_DIM;
    int k  = r % IN_DIM;
    const float* W = (m3 == 0) ? Wq : (m3 == 1) ? Wk : Wv;
    Wt[i] = (_Float16)W[k * HID + n];                // Wt[m3][n][k] = W[k][n]
}

// -------------------------------------------------------- QKV GEMM via WMMA
// One wave per (16-row node tile, matrix m3). A resides in registers across
// the 8 N-tile loop; 4 x v_wmma_f32_16x16x32_f16 per tile (K=128).
__global__ void __launch_bounds__(256)
qkv_gemm_kernel(const float* __restrict__ x,
                const _Float16* __restrict__ Wt,
                const float* __restrict__ bq,
                const float* __restrict__ bk,
                const float* __restrict__ bv,
                _Float16* __restrict__ Qo,
                _Float16* __restrict__ Ko,
                _Float16* __restrict__ Vo) {
    int wave = (blockIdx.x * blockDim.x + threadIdx.x) >> 5;
    int lane = threadIdx.x & 31;
    const int nWaves = (N_NODES / 16) * 3;
    if (wave >= nWaves) return;                      // wave-uniform exit (EXEC all-1)

    int m3    = wave % 3;
    int mTile = wave / 3;
    int lo    = lane & 15;
    int hi    = lane >> 4;                           // 0: lanes 0-15, 1: lanes 16-31

    // --- A tile: rows mTile*16+lo of x, f32 -> f16, ISA 16-bit A 16x32 layout.
    // lanes 0-15 hold K = kb*32 + {0..7, 16..23}; lanes 16-31 the +8 variants.
    int row = mTile * 16 + lo;
    const float4* xr = (const float4*)(x + (size_t)row * IN_DIM);
    v16h a[4];
#pragma unroll
    for (int kb = 0; kb < 4; ++kb) {
        float4 p0 = xr[kb * 8 + hi * 2];
        float4 p1 = xr[kb * 8 + hi * 2 + 1];
        float4 p2 = xr[kb * 8 + 4 + hi * 2];
        float4 p3 = xr[kb * 8 + 4 + hi * 2 + 1];
        AV t;
        t.e[0]  = (_Float16)p0.x; t.e[1]  = (_Float16)p0.y;
        t.e[2]  = (_Float16)p0.z; t.e[3]  = (_Float16)p0.w;
        t.e[4]  = (_Float16)p1.x; t.e[5]  = (_Float16)p1.y;
        t.e[6]  = (_Float16)p1.z; t.e[7]  = (_Float16)p1.w;
        t.e[8]  = (_Float16)p2.x; t.e[9]  = (_Float16)p2.y;
        t.e[10] = (_Float16)p2.z; t.e[11] = (_Float16)p2.w;
        t.e[12] = (_Float16)p3.x; t.e[13] = (_Float16)p3.y;
        t.e[14] = (_Float16)p3.z; t.e[15] = (_Float16)p3.w;
        a[kb] = t.v;
    }

    const float* bias  = (m3 == 0) ? bq : (m3 == 1) ? bk : bv;
    _Float16*    out   = (m3 == 0) ? Qo : (m3 == 1) ? Ko : Vo;

#pragma unroll
    for (int nt = 0; nt < 8; ++nt) {
        int col = nt * 16 + lo;
        // B layout: lanes 0-15 hold K = kb*32 + {0..7, 16..23} of column N=lo,
        //           lanes 16-31 the +8 variants. Wt is [n][k] -> contiguous.
        const v8h* wc = (const v8h*)(Wt + ((size_t)m3 * HID + col) * IN_DIM);
        v8f c = {};
#pragma unroll
        for (int kb = 0; kb < 4; ++kb) {
            BV b;
            b.h8[0] = wc[kb * 4 + hi];
            b.h8[1] = wc[kb * 4 + 2 + hi];
            c = __builtin_amdgcn_wmma_f32_16x16x32_f16(
                    false, a[kb], false, b.v, (short)0, c, false, false);
        }
        // C/D layout: VGPR r -> row M = r + hi*8, col N = lo.
        float bb = bias[col];
#pragma unroll
        for (int r = 0; r < 8; ++r) {
            int node = mTile * 16 + r + hi * 8;
            out[(size_t)node * HID + col] = (_Float16)(c[r] + bb);
        }
    }
}

// ----------------------------------------------------- edge gather / scatter
// One wave per edge; lane covers 4 of the 128 (head,dim) slots.
// Head h = lane>>2 lives in lanes 4h..4h+3 -> shfl_xor(1),shfl_xor(2) reduce.
__global__ void __launch_bounds__(256)
edge_agg_kernel(const int* __restrict__ src,
                const int* __restrict__ dst,
                const _Float16* __restrict__ Q,
                const _Float16* __restrict__ K,
                const _Float16* __restrict__ V,
                float* __restrict__ wV) {
    int lane = threadIdx.x & 31;
    int wid  = (blockIdx.x * blockDim.x + threadIdx.x) >> 5;
    int nw   = (gridDim.x * blockDim.x) >> 5;
    for (int e = wid; e < N_EDGES; e += nw) {
        int s = src[e];
        int d = dst[e];
        V4U kh, qh, vh;
        kh.v = *((const v4h*)(K + (size_t)s * HID) + lane);
        qh.v = *((const v4h*)(Q + (size_t)d * HID) + lane);
        vh.v = *((const v4h*)(V + (size_t)s * HID) + lane);
        float part = (float)kh.e[0] * (float)qh.e[0]
                   + (float)kh.e[1] * (float)qh.e[1]
                   + (float)kh.e[2] * (float)qh.e[2]
                   + (float)kh.e[3] * (float)qh.e[3];
        part += __shfl_xor(part, 1);
        part += __shfl_xor(part, 2);
        float sc = part * 0.25f;                 // 1/sqrt(OUT_DIM)
        sc = fminf(5.0f, fmaxf(-5.0f, sc));
        float w = __expf(sc);
        float* o = wV + (size_t)d * HID + lane * 4;
        unsafeAtomicAdd(o + 0, w * (float)vh.e[0]);
        unsafeAtomicAdd(o + 1, w * (float)vh.e[1]);
        unsafeAtomicAdd(o + 2, w * (float)vh.e[2]);
        unsafeAtomicAdd(o + 3, w * (float)vh.e[3]);
    }
}

extern "C" void kernel_launch(void* const* d_in, const int* in_sizes, int n_in,
                              void* d_out, int out_size, void* d_ws, size_t ws_size,
                              hipStream_t stream) {
    (void)in_sizes; (void)n_in; (void)out_size; (void)ws_size;
    const float* x   = (const float*)d_in[0];
    const int*   src = (const int*)d_in[1];
    const int*   dst = (const int*)d_in[2];
    const float* Wq  = (const float*)d_in[3];
    const float* bq  = (const float*)d_in[4];
    const float* Wk  = (const float*)d_in[5];
    const float* bk  = (const float*)d_in[6];
    const float* Wv  = (const float*)d_in[7];
    const float* bv  = (const float*)d_in[8];
    float* out = (float*)d_out;

    // Workspace layout (bytes):
    //   Wt f16 [3][128][128]             @ 0        (98304 B)
    //   Q  f16 [100000][128]             @ 98304    (25.6 MB)
    //   K  f16                           @ 98304 + 25600000
    //   V  f16                           @ 98304 + 51200000
    char* ws = (char*)d_ws;
    _Float16* Wt = (_Float16*)ws;
    _Float16* Q  = (_Float16*)(ws + 98304);
    _Float16* K  = (_Float16*)(ws + 98304 + 25600000);
    _Float16* V  = (_Float16*)(ws + 98304 + 51200000);

    // 1) zero wV (harness poisons d_out)
    int n4 = (N_NODES * HID) / 4;                        // 3200000 float4
    zero_out_kernel<<<(n4 + 255) / 256, 256, 0, stream>>>((float4*)out, n4);

    // 2) transpose/convert weights to f16 [n][k]
    prep_w_kernel<<<(3 * HID * IN_DIM + 255) / 256, 256, 0, stream>>>(Wq, Wk, Wv, Wt);

    // 3) QKV projection via WMMA: one wave per (16-node tile, matrix)
    int nWaves = (N_NODES / 16) * 3;                     // 18750
    int blocks = (nWaves * 32 + 255) / 256;
    qkv_gemm_kernel<<<blocks, 256, 0, stream>>>(x, Wt, bq, bk, bv, Q, K, V);

    // 4) edge phase: gather-dot, exp, atomic scatter
    edge_agg_kernel<<<8192, 256, 0, stream>>>(src, dst, Q, K, V, out);
}
